// RegionProposalNetwork_39522289057800
// MI455X (gfx1250) — matrix operand order repaired and checked by reading, not compiled, not run on hardware
//
#include <hip/hip_runtime.h>
#include <hip/hip_bf16.h>

typedef __attribute__((ext_vector_type(16))) _Float16 v16h;
typedef __attribute__((ext_vector_type(8)))  float     v8f;
typedef unsigned __attribute__((ext_vector_type(4)))   v4u;
typedef int      __attribute__((ext_vector_type(4)))   v4i;
typedef int      __attribute__((ext_vector_type(8)))   v8i;

#define NANCH  360000           // 200*200*9
#define PRE    2000
#define CAP    2048
#define POST   1000
#define NEGV   (-1e9f)
#define CLIPV  4.135166556742356f   // log(1000/16)

// channel -> LDS position permutation so that one contiguous 16-half read
// yields the WMMA f16 A-fragment K order ({0..7,16..23} or {8..15,24..31}).
__device__ __forceinline__ int permch(int ch) {
  return (ch & 7) | ((ch & 8) << 1) | ((ch & 16) >> 1);
}

// Low 32 bits of a generic pointer into LDS = LDS byte offset (aperture map).
__device__ __forceinline__ unsigned lds_off(const void* p) {
  return (unsigned)(unsigned long long)(uintptr_t)p;
}

// Issue a 1-D TDM copy: n8 units of 8 bytes, global -> LDS.
// D# group0: count=1 | lds_addr | global_addr | type=2 ("image").
// D# group1: data_size=3 (8B), tensor_dim0=tile_dim0=n8, dims1..=1.
__device__ __forceinline__ void tdm_load_1d(const void* gsrc, unsigned lds_addr,
                                            unsigned n8) {
  unsigned long long ga = (unsigned long long)(uintptr_t)gsrc;
  v4u g0;
  g0.x = 1u;                                        // count=1, user descriptor
  g0.y = lds_addr;                                  // LDS byte address
  g0.z = (unsigned)(ga & 0xffffffffu);              // global_addr[31:0]
  g0.w = (unsigned)((ga >> 32) & 0x01ffffffu) | (2u << 30);  // [56:32] | type=2
  v8i g1;
  g1[0] = 0x30000;                 // workgroup_mask=0, data_size=3 (8 bytes)
  g1[1] = (int)(n8 << 16);         // tensor_dim0[15:0] in [31:16]
  g1[2] = (int)((n8 >> 16) & 0xffffu) | (1 << 16);  // td0[31:16] | tensor_dim1=1
  g1[3] = (int)(n8 << 16);         // tensor_dim1 hi=0 | tile_dim0=n8
  g1[4] = 1;                       // tile_dim1=1, tile_dim2=0
  g1[5] = (int)n8;                 // tensor_dim0_stride (unused for 1 row)
  g1[6] = 0;
  g1[7] = 0;
  v4i z4 = {0, 0, 0, 0};
#if __clang_major__ >= 23
  v8i z8 = {0, 0, 0, 0, 0, 0, 0, 0};
  __builtin_amdgcn_tensor_load_to_lds(g0, g1, z4, z4, z8, 0);
#else
  __builtin_amdgcn_tensor_load_to_lds(g0, g1, z4, z4, 0);
#endif
}

// ---------------- weight prepack: B-fragment layout, f16 -------------------
// B 32x16 f16 fragment (ISA 7.12.2 pattern): lane 0-15 -> N=lane, K={0..7,16..23};
// lane 16-31 -> N=lane-16, K={8..15,24..31}.  One v16h per lane per fragment.
__global__ __launch_bounds__(256)
void prepack_conv(const float* __restrict__ cw, _Float16* __restrict__ wq)
{
  int g = blockIdx.x * 256 + threadIdx.x;
  if (g >= 72 * 16 * 32 * 16) return;
  int hidx  = g & 15;          // element within lane (0..15)
  int lane  = (g >> 4) & 31;
  int nt    = (g >> 9) & 15;   // 16 N-tiles of 16 -> 256 out channels
  int chunk = g >> 13;         // chunk = tap*8 + icc  (K-chunk of 32)
  int k   = ((lane >= 16) ? 8 : 0) + (hidx & 7) + ((hidx & 8) << 1);
  int tap = chunk >> 3, icc = chunk & 7;
  int oc  = nt * 16 + (lane & 15);
  int ic  = icc * 32 + k;
  wq[g] = (_Float16)cw[(oc * 256 + ic) * 9 + tap];
}

__global__ __launch_bounds__(256)
void prepack_head(const float* __restrict__ cls_w, const float* __restrict__ bbox_w,
                  _Float16* __restrict__ hq)
{
  int g = blockIdx.x * 256 + threadIdx.x;
  if (g >= 8 * 3 * 32 * 16) return;
  int hidx = g & 15;
  int lane = (g >> 4) & 31;
  int cn   = g >> 9;           // icc*3 + nt  (3 N-tiles cover 48 >= 45 outs)
  int nt = cn % 3, icc = cn / 3;
  int k = ((lane >= 16) ? 8 : 0) + (hidx & 7) + ((hidx & 8) << 1);
  int j = nt * 16 + (lane & 15);
  int c = icc * 32 + k;
  float val = 0.f;
  if (j < 9)       val = cls_w[j * 256 + c];
  else if (j < 45) val = bbox_w[(j - 9) * 256 + c];
  hq[g] = (_Float16)val;
}

// ------------- fused 3x3 conv + ReLU + cls/bbox heads + decode -------------
__global__ __launch_bounds__(256)
void rpn_conv_head(const float* __restrict__ feat,
                   const float* __restrict__ conv_b,
                   const float* __restrict__ cls_b,
                   const float* __restrict__ bbox_b,
                   const _Float16* __restrict__ wq,
                   const _Float16* __restrict__ hq,
                   float* __restrict__ props,
                   float* __restrict__ scores)
{
  __shared__ __align__(64) unsigned char smem[57344];
  _Float16* As = (_Float16*)smem;             // [3][66][32ch perm]  12,672 B
  // weight chunk double buffer: smem+16384 (buf 0) / smem+32768 (buf 1)

  const int tid  = threadIdx.x;
  const int lane = tid & 31;
  const int wave = tid >> 5;
  const int msub  = wave >> 1;    // 4 M-subtiles of 16 rows -> 64 cols strip
  const int nhalf = wave & 1;     // 2 N-halves of 128 out-channels
  const int blk = blockIdx.x;
  const int b = blk / 800;
  const int rem = blk % 800;
  const int h = rem >> 2;
  const int wbase = (rem & 3) * 64;

  v8f acc[8] = {};
  const int mrow = msub * 16 + (lane & 15);
  const int coff = lane & 16;     // 0 or 16: which permuted half-chunk

  const unsigned bsoff0 = lds_off(smem + 16384);

  // Prologue: TDM-prefetch weight chunk 0 (tap 0, icc 0) into buffer 0.
  if (wave == 0) tdm_load_1d(wq, bsoff0, 2048);

  int it = 0;
  for (int icc = 0; icc < 8; ++icc) {
    // stage 32 channels x 3 rows x 66 cols of features, f32 -> f16, permuted
    for (int e = tid; e < 3 * 66 * 32; e += 256) {
      int ch = e / 198;
      int r2 = e - ch * 198;
      int rr = r2 / 66;
      int cc = r2 - rr * 66;
      int hh = h + rr - 1;
      int ww = wbase + cc - 1;
      float v = 0.f;
      if (hh >= 0 && hh < 200 && ww >= 0 && ww < 200)
        v = feat[(((b * 256) + (icc * 32 + ch)) * 200 + hh) * 200 + ww];
      As[(rr * 66 + cc) * 32 + permch(ch)] = (_Float16)v;
    }
    for (int tap = 0; tap < 9; ++tap, ++it) {
      const int p = it & 1;
      if (wave == 0) __builtin_amdgcn_s_wait_tensorcnt((short)0);
      __syncthreads();            // releases buffer p data + As to all waves
      // TDM-prefetch next weight chunk into the other buffer (overlaps WMMA)
      int ntap = tap + 1, nicc = icc;
      if (ntap == 9) { ntap = 0; nicc = icc + 1; }
      if (wave == 0 && nicc < 8)
        tdm_load_1d(wq + (size_t)(ntap * 8 + nicc) * 8192,
                    bsoff0 + (unsigned)((p ^ 1) << 14), 2048);

      const int kh = tap / 3, kw = tap - kh * 3;
      v16h a = *(const v16h*)&As[((kh * 66) + (mrow + kw)) * 32 + coff];
      // NOTE: derive buffer pointer from the __shared__ base so clang keeps
      // the addrspace(3) provenance and emits ds_load_b128 (not flat_load).
      const _Float16* bs = (const _Float16*)(smem + 16384 + (p << 14));
#pragma unroll
      for (int nt = 0; nt < 8; ++nt) {
        v16h bf = *(const v16h*)&bs[(((nhalf * 8 + nt) * 32) + lane) * 16];
        acc[nt] = __builtin_amdgcn_wmma_f32_16x16x32_f16(
            false, a, false, bf, (short)0, acc[nt], false, false);
      }
    }
    __syncthreads();              // retire As reads before restaging
  }

  // bias + ReLU; restage t (f16, permuted) for the head GEMM.
  // Overlap: TDM the head weights (24 KB) into smem+32768 meanwhile.
  _Float16* hw = (_Float16*)(smem + 32768);   // [8][3][32][16]  24,576 B
  if (wave == 0) tdm_load_1d(hq, lds_off(hw), 3072);
  _Float16* ts = (_Float16*)smem;             // [64 m][256 ch perm]  32,768 B
#pragma unroll
  for (int nt = 0; nt < 8; ++nt) {
    const int oc = (nhalf * 8 + nt) * 16 + (lane & 15);
    const float cb = conv_b[oc];
    const int cpos = (oc & ~31) | permch(oc & 31);
#pragma unroll
    for (int r = 0; r < 8; ++r) {
      int m = msub * 16 + r + ((lane & 16) ? 8 : 0);
      ts[m * 256 + cpos] = (_Float16)fmaxf(acc[nt][r] + cb, 0.f);
    }
  }
  if (wave == 0) __builtin_amdgcn_s_wait_tensorcnt((short)0);
  __syncthreads();

  // head GEMM: [64 x 256] @ [256 x 48]
  v8f hacc[2] = {};
  for (int icc = 0; icc < 8; ++icc) {
    v16h a = *(const v16h*)&ts[mrow * 256 + icc * 32 + coff];
#pragma unroll
    for (int q = 0; q < 2; ++q) {
      const int nt = nhalf * 2 + q;
      if (nt < 3) {      // wave-uniform -> EXEC stays all-ones around WMMA
        v16h bf = *(const v16h*)&hw[(((icc * 3 + nt) * 32) + lane) * 16];
        hacc[q] = __builtin_amdgcn_wmma_f32_16x16x32_f16(
            false, a, false, bf, (short)0, hacc[q], false, false);
      }
    }
  }
  __syncthreads();
  float* hout = (float*)smem;                 // [64][48]  12,288 B (alias ts)
#pragma unroll
  for (int q = 0; q < 2; ++q) {
    const int nt = nhalf * 2 + q;
    if (nt < 3) {
      const int j = nt * 16 + (lane & 15);
      float hb = (j < 9) ? cls_b[j] : ((j < 45) ? bbox_b[j - 9] : 0.f);
#pragma unroll
      for (int r = 0; r < 8; ++r) {
        int m = msub * 16 + r + ((lane & 16) ? 8 : 0);
        hout[m * 48 + j] = hacc[q][r] + hb;
      }
    }
  }
  __syncthreads();

  // decode proposals (anchors inline; repo swaps h/w ratios, scale-major)
  const float sizes_c[3]  = {32.f, 64.f, 128.f};
  const float ratios_c[3] = {0.5f, 1.f, 2.f};
  for (int e = tid; e < 64 * 9; e += 256) {
    int m = e / 9, a = e - m * 9;
    int w = wbase + m;
    if (w < 200) {
      int si = a / 3, ri = a - si * 3;
      float sq  = sqrtf(ratios_c[ri]);
      float wsz = sizes_c[si] * sq;       // ws = scale * sqrt(ar)
      float hsz = sizes_c[si] / sq;       // hs = scale / sqrt(ar)
      float cxa = 4.f * (float)w, cya = 4.f * (float)h;
      const float* hb = &hout[m * 48 + 9 + a * 4];
      float dx = hb[0], dy = hb[1];
      float dw = fminf(hb[2], CLIPV), dh = fminf(hb[3], CLIPV);
      float cx = dx * wsz + cxa, cy = dy * hsz + cya;
      float pw = expf(dw) * wsz,  ph = expf(dh) * hsz;
      int n = (h * 200 + w) * 9 + a;
      float* pp = props + ((size_t)b * NANCH + n) * 4;
      pp[0] = cx - 0.5f * pw; pp[1] = cy - 0.5f * ph;
      pp[2] = cx + 0.5f * pw; pp[3] = cy + 0.5f * ph;
      scores[(size_t)b * NANCH + n] = hout[m * 48 + a];
    }
  }
}

// ---------------- exact top-2000 via 3-pass radix select + bitonic ---------
__device__ __forceinline__ unsigned fkey(float f) {
  unsigned u = __float_as_uint(f);
  return (u & 0x80000000u) ? ~u : (u | 0x80000000u);   // monotone float key
}

__global__ __launch_bounds__(256)
void rpn_topk(const float* __restrict__ scores,
              const float* __restrict__ props,
              float* __restrict__ tk_scores,
              float* __restrict__ tk_boxes)
{
  __shared__ unsigned hist[2048];
  __shared__ unsigned long long key64[CAP];
  __shared__ unsigned sh_pref, sh_tgt, sh_cnt;
  const int b = blockIdx.x;
  const int tid = threadIdx.x;
  const float* sc = scores + (size_t)b * NANCH;

  unsigned pref = 0, target = PRE;
  const int shifts[3] = {21, 10, 0};
  const int widths[3] = {11, 11, 10};
  for (int p = 0; p < 3; ++p) {
    const int sh = shifts[p], wdt = widths[p];
    const unsigned nb = 1u << wdt;
    for (int i = tid; i < 2048; i += 256) hist[i] = 0;
    __syncthreads();
    const int hb = sh + wdt;
    for (int i = tid; i < NANCH; i += 256) {
      unsigned key = fkey(sc[i]);
      unsigned hi = (hb >= 32) ? 0u : (key >> hb);
      if (hi == pref) atomicAdd(&hist[(key >> sh) & (nb - 1)], 1u);
    }
    __syncthreads();
    if (tid == 0) {
      unsigned cum = 0;
      int bin = (int)nb - 1;
      for (; bin > 0; --bin) {
        unsigned c = hist[bin];
        if (cum + c >= target) break;
        cum += c;
      }
      sh_pref = (pref << wdt) | (unsigned)bin;
      sh_tgt  = target - cum;
    }
    __syncthreads();
    pref = sh_pref; target = sh_tgt;
    __syncthreads();
  }
  const unsigned T = pref;   // exact 32-bit threshold key -> deterministic set
  if (tid == 0) sh_cnt = 0;
  __syncthreads();
  for (int i = tid; i < NANCH; i += 256) {
    unsigned key = fkey(sc[i]);
    if (key >= T) {
      unsigned pos = atomicAdd(&sh_cnt, 1u);
      if (pos < CAP)
        key64[pos] = (((unsigned long long)(~key)) << 32) | (unsigned)i;
    }
  }
  __syncthreads();
  const unsigned cnt = sh_cnt < CAP ? sh_cnt : CAP;
  for (int r = tid; r < CAP; r += 256)
    if ((unsigned)r >= cnt) key64[r] = ~0ull;
  // bitonic sort ascending => descending score, index tie-break
  for (unsigned k = 2; k <= CAP; k <<= 1) {
    for (unsigned j = k >> 1; j > 0; j >>= 1) {
      __syncthreads();
      for (unsigned t = tid; t < CAP; t += 256) {
        unsigned ixj = t ^ j;
        if (ixj > t) {
          unsigned long long x = key64[t], y = key64[ixj];
          bool up = ((t & k) == 0);
          if ((x > y) == up) { key64[t] = y; key64[ixj] = x; }
        }
      }
    }
  }
  __syncthreads();
  for (int r = tid; r < CAP; r += 256) {
    unsigned long long kk = key64[r];
    float s = NEGV, x1 = 0, y1 = 0, x2 = 0, y2 = 0;
    if (r < PRE && kk != ~0ull) {
      int idx = (int)(kk & 0xffffffffu);
      s = sc[idx];
      const float* pp = props + ((size_t)b * NANCH + idx) * 4;
      x1 = fminf(fmaxf(pp[0], 0.f), 800.f);
      y1 = fminf(fmaxf(pp[1], 0.f), 800.f);
      x2 = fminf(fmaxf(pp[2], 0.f), 800.f);
      y2 = fminf(fmaxf(pp[3], 0.f), 800.f);
      if (!((x2 - x1) >= 1e-3f && (y2 - y1) >= 1e-3f)) s = NEGV;
    }
    tk_scores[b * CAP + r] = s;
    float* ob = tk_boxes + (size_t)(b * CAP + r) * 4;
    ob[0] = x1; ob[1] = y1; ob[2] = x2; ob[3] = y2;
  }
}

// ---------------------------- greedy NMS -----------------------------------
__global__ __launch_bounds__(256)
void rpn_nms(const float* __restrict__ tk_scores,
             const float* __restrict__ tk_boxes,
             int* __restrict__ keep)
{
  __shared__ float bx[CAP * 4];
  __shared__ int kp[CAP];
  const int b = blockIdx.x;
  const int tid = threadIdx.x;
  for (int r = tid; r < CAP; r += 256) {
    kp[r] = tk_scores[b * CAP + r] > (NEGV * 0.5f) ? 1 : 0;
    const float* pp = tk_boxes + (size_t)(b * CAP + r) * 4;
    bx[r*4+0]=pp[0]; bx[r*4+1]=pp[1]; bx[r*4+2]=pp[2]; bx[r*4+3]=pp[3];
  }
  __syncthreads();
  for (int i = 0; i < CAP; ++i) {
    __syncthreads();
    if (!kp[i]) continue;
    float x1 = bx[i*4], y1 = bx[i*4+1], x2 = bx[i*4+2], y2 = bx[i*4+3];
    float ai = (x2 - x1) * (y2 - y1);
    for (int j = i + 1 + tid; j < CAP; j += 256) {
      if (!kp[j]) continue;
      float xx1 = fmaxf(x1, bx[j*4+0]);
      float yy1 = fmaxf(y1, bx[j*4+1]);
      float xx2 = fminf(x2, bx[j*4+2]);
      float yy2 = fminf(y2, bx[j*4+3]);
      float inter = fmaxf(xx2 - xx1, 0.f) * fmaxf(yy2 - yy1, 0.f);
      float aj = (bx[j*4+2] - bx[j*4+0]) * (bx[j*4+3] - bx[j*4+1]);
      float iou = inter / (ai + aj - inter + 1e-12f);
      if (iou > 0.7f) kp[j] = 0;
    }
  }
  __syncthreads();
  for (int r = tid; r < CAP; r += 256) keep[b * CAP + r] = kp[r];
}

// --------- compaction (kept first, suppressed after, both in order) --------
__global__ __launch_bounds__(256)
void rpn_finalize(const float* __restrict__ tk_scores,
                  const float* __restrict__ tk_boxes,
                  const int* __restrict__ keep,
                  float* __restrict__ out)
{
  __shared__ int part[256];
  const int b = blockIdx.x;
  const int tid = threadIdx.x;
  int loc[8];
  int s = 0;
  const int r0 = tid * 8;
  for (int q = 0; q < 8; ++q) { loc[q] = s; s += keep[b * CAP + r0 + q]; }
  part[tid] = s;
  __syncthreads();
  for (int off = 1; off < 256; off <<= 1) {
    int v = (tid >= off) ? part[tid - off] : 0;
    __syncthreads();
    part[tid] += v;
    __syncthreads();
  }
  const int total = part[255];
  const int excl = (tid == 0) ? 0 : part[tid - 1];
  for (int q = 0; q < 8; ++q) {
    const int r = r0 + q;
    const int kf = keep[b * CAP + r];
    const int pre = excl + loc[q];
    const int rank = kf ? pre : total + (r - pre);
    if (rank < POST) {
      const float* pp = tk_boxes + (size_t)(b * CAP + r) * 4;
      float* o = out + (size_t)(b * POST + rank) * 5;
      o[0] = pp[0]; o[1] = pp[1]; o[2] = pp[2]; o[3] = pp[3];
      o[4] = kf ? tk_scores[b * CAP + r] : NEGV;
    }
  }
}

extern "C" void kernel_launch(void* const* d_in, const int* in_sizes, int n_in,
                              void* d_out, int out_size, void* d_ws, size_t ws_size,
                              hipStream_t stream)
{
  const float* feat   = (const float*)d_in[0];
  const float* conv_w = (const float*)d_in[1];
  const float* conv_b = (const float*)d_in[2];
  const float* cls_w  = (const float*)d_in[3];
  const float* cls_b  = (const float*)d_in[4];
  const float* bbox_w = (const float*)d_in[5];
  const float* bbox_b = (const float*)d_in[6];
  float* out = (float*)d_out;

  char* ws = (char*)d_ws;
  _Float16* wq = (_Float16*)ws;                        // 1,179,648 B
  _Float16* hq = (_Float16*)(ws + 1179648);            //    24,576 B
  float* props = (float*)(ws + 1179648 + 24576);       // 23,040,000 B
  float* scores    = props + (size_t)4 * NANCH * 4;    //  5,760,000 B
  float* tk_scores = scores + (size_t)4 * NANCH;       //    32,768 B
  float* tk_boxes  = tk_scores + 4 * CAP;              //   131,072 B
  int*   keep      = (int*)(tk_boxes + 4 * CAP * 4);   //    32,768 B

  prepack_conv<<<2304, 256, 0, stream>>>(conv_w, wq);
  prepack_head<<<48, 256, 0, stream>>>(cls_w, bbox_w, hq);
  rpn_conv_head<<<3200, 256, 0, stream>>>(feat, conv_b, cls_b, bbox_b,
                                          wq, hq, props, scores);
  rpn_topk<<<4, 256, 0, stream>>>(scores, props, tk_scores, tk_boxes);
  rpn_nms<<<4, 256, 0, stream>>>(tk_scores, tk_boxes, keep);
  rpn_finalize<<<4, 256, 0, stream>>>(tk_scores, tk_boxes, keep, out);
  (void)in_sizes; (void)n_in; (void)out_size; (void)ws_size;
}
